// GemNetOCForceHead_55009941128039
// MI455X (gfx1250) — matrix operand order repaired and checked by reading, not compiled, not run on hardware
//
#include <hip/hip_runtime.h>
#include <stdint.h>

// ---------------------------------------------------------------------------
// GemNet-OC force head, fused, bf16 WMMA (f32 accumulate) for gfx1250.
// Round 4: sched_group_barrier-pinned software pipeline (depth-3 B rotation)
// so ds_read_b128 runs 3 fragments ahead of each v_wmma.
// ---------------------------------------------------------------------------

typedef __attribute__((ext_vector_type(16))) __bf16 v16bf;
typedef __attribute__((ext_vector_type(8)))  float  v8f;

#define INV_SQRT2f 0.70710678118654752440f
#define SSCALEf    (1.0f / 0.6f)

static constexpr int Dm     = 256;     // emb size
static constexpr int KIN    = 1280;    // D * NB
static constexpr int MT     = 128;     // edges per workgroup
static constexpr int NT     = 16;      // 16-wide N tiles (16*16 = 256)
static constexpr int PITCH  = 264;     // bf16 elems per LDS row (256 + 8 pad)
static constexpr int APITCH = 72;      // bf16 elems per A-staging row (64 + 8)
static constexpr int SPITCH = 260;     // f32 elems per skip row (256 + 4 pad)

// bf16 weight layout inside d_ws (uint16 element offsets)
static constexpr int WIN_OFF  = 0;
static constexpr int WIN_SZ   = KIN * Dm;            // 327680
static constexpr int WR_SZ    = Dm * Dm;             // 65536
static constexpr int WR1A_OFF = WIN_OFF + WIN_SZ;
static constexpr int WR1B_OFF = WR1A_OFF + WR_SZ;
static constexpr int WR2A_OFF = WR1B_OFF + WR_SZ;
static constexpr int WR2B_OFF = WR2A_OFF + WR_SZ;
static constexpr int WTOT     = WR2B_OFF + WR_SZ;    // 589824 bf16 (~1.2 MB)

__device__ __forceinline__ uint32_t pk_bf16(float a, float b) {
  union { float f; uint32_t u; } ua, ub;
  ua.f = a; ub.f = b;
  uint32_t x = ua.u + (0x7FFFu + ((ua.u >> 16) & 1u));
  uint32_t y = ub.u + (0x7FFFu + ((ub.u >> 16) & 1u));
  return (x >> 16) | (y & 0xFFFF0000u);
}

__device__ __forceinline__ uint16_t bf16_1(float a) {
  union { float f; uint32_t u; } ua; ua.f = a;
  return (uint16_t)((ua.u + (0x7FFFu + ((ua.u >> 16) & 1u))) >> 16);
}

__device__ __forceinline__ float ssilu(float x) {
  return (x / (1.0f + __expf(-x))) * SSCALEf;
}

union AFrag { v16bf v; uint4 q[2]; };
union BFrag { v16bf v; uint4 q[2]; };
union Acc   { v8f  v; float f[8]; };

__device__ __forceinline__ v8f wmma_bf16(const AFrag& a, const BFrag& b, v8f c) {
  return __builtin_amdgcn_wmma_f32_16x16x32_bf16(
      /*neg_a=*/false, a.v, /*neg_b=*/false, b.v,
      /*c_mod=*/(short)0, c, /*reuse_a=*/false, /*reuse_b=*/false);
}

// --- CDNA5 async global->LDS copy (16B per lane), tracked by ASYNCcnt -------
__device__ __forceinline__ void async_cp16(void* ldsdst, const void* gsrc) {
  uint32_t lds_off = (uint32_t)(uintptr_t)ldsdst;   // addr[31:0] = LDS offset
  asm volatile("global_load_async_to_lds_b128 %0, %1, off"
               :: "v"(lds_off), "v"(gsrc)
               : "memory");
}
__device__ __forceinline__ void wait_async0() {
  asm volatile("s_wait_asynccnt 0x0" ::: "memory");
}

// --- one staged 64-wide K chunk: 32 WMMAs, B pipelined at depth 3 -----------
// Region contents: 68 ds_read_b128 (4 A + 64 B) and 32 v_wmma.
__device__ __forceinline__ void mma_64k(const uint16_t* __restrict__ aRow,
                                        const uint16_t* __restrict__ wbuf,
                                        int lane, int h, Acc* acc) {
  AFrag a[2];
  const uint16_t* ab = aRow + 8 * h;
  a[0].q[0] = *(const uint4*)(ab);
  a[0].q[1] = *(const uint4*)(ab + 16);
  a[1].q[0] = *(const uint4*)(ab + 32);
  a[1].q[1] = *(const uint4*)(ab + 48);

  const uint16_t* wb0 = wbuf + lane * PITCH;          // K rows 0..31
  const uint16_t* wb1 = wbuf + (32 + lane) * PITCH;   // K rows 32..63

  BFrag b[4];
#pragma unroll
  for (int t = 0; t < 3; ++t) {                       // preload 3 fragments
    b[t].q[0] = *(const uint4*)(wb0 + 16 * t);
    b[t].q[1] = *(const uint4*)(wb0 + 16 * t + 8);
  }

#pragma unroll
  for (int s = 0; s < 32; ++s) {                      // flattened 2x16 steps
    int t = s + 3;
    if (t < 32) {
      const uint16_t* wb = (t < 16) ? wb0 : wb1;
      int tn = t & 15;
      b[t & 3].q[0] = *(const uint4*)(wb + 16 * tn);
      b[t & 3].q[1] = *(const uint4*)(wb + 16 * tn + 8);
    }
    acc[s & 15].v = wmma_bf16(a[s >> 4], b[s & 3], acc[s & 15].v);
  }

  // pin the pipeline: 10 ds_read up front, then 1 wmma : 2 ds_read, 3-deep
  __builtin_amdgcn_sched_group_barrier(0x100, 10, 0);   // 4 A + 6 B reads
#pragma unroll
  for (int s = 0; s < 29; ++s) {
    __builtin_amdgcn_sched_group_barrier(0x008, 1, 0);  // 1 WMMA
    __builtin_amdgcn_sched_group_barrier(0x100, 2, 0);  // next B fragment
  }
  __builtin_amdgcn_sched_group_barrier(0x008, 3, 0);    // drain: last 3 WMMA
}

// ---------------------------------------------------------------------------
// Prologue kernels
// ---------------------------------------------------------------------------
__global__ void zero_out_kernel(float* out, int n) {
  int i = blockIdx.x * blockDim.x + threadIdx.x;
  if (i < n) out[i] = 0.0f;
}

__global__ void convert_weights_kernel(const float* __restrict__ win,
                                       const float* __restrict__ r1a,
                                       const float* __restrict__ r1b,
                                       const float* __restrict__ r2a,
                                       const float* __restrict__ r2b,
                                       uint16_t* __restrict__ ws) {
  int i = blockIdx.x * blockDim.x + threadIdx.x;   // bf16-pair index
  if (i >= WTOT / 2) return;
  int e = i * 2;
  const float* src; int off;
  if      (e < WR1A_OFF) { src = win; off = e - WIN_OFF;  }
  else if (e < WR1B_OFF) { src = r1a; off = e - WR1A_OFF; }
  else if (e < WR2A_OFF) { src = r1b; off = e - WR1B_OFF; }
  else if (e < WR2B_OFF) { src = r2a; off = e - WR2A_OFF; }
  else                   { src = r2b; off = e - WR2B_OFF; }
  int pr = (off >> 8) & 63, pc = off & 255, ps = pc & 15;
  int pk = 32 * (pr >> 5) + 8 * ((pr >> 4) & 1) + (ps < 8 ? ps : 8 + ps);
  int pn = (pc & ~15) + (pr & 15);
  int so = ((off >> 14) * 64 + pk) * Dm + pn;
  ((uint32_t*)ws)[i] = pk_bf16(src[so], src[so + Dm]);
}

// ---------------------------------------------------------------------------
// Fused force-head kernel: 256 threads = 8 waves, 128 edges per workgroup.
// ---------------------------------------------------------------------------
__launch_bounds__(256, 1)
__global__ void force_head_kernel(const float* __restrict__ x_cat,
                                  const float* __restrict__ edge_vec,
                                  const int*   __restrict__ edge_idx,
                                  const uint16_t* __restrict__ wbf,
                                  const float* __restrict__ w_out,
                                  float* __restrict__ forces,
                                  int e_tot) {
  extern __shared__ float4 smem4[];
  uint16_t* xh   = (uint16_t*)smem4;            // [128][264] bf16 activations
  uint16_t* wbuf = xh   + MT * PITCH;           // [64][264]  bf16 weight chunk
  uint16_t* abuf = wbuf + 64 * PITCH;           // [8][16][72] bf16 A staging
  float*    wout = (float*)(abuf + 8 * 16 * APITCH); // [256]
  float*    fst  = wout + Dm;                   // [128]
  float*    sbuf = fst + MT;                    // [128][260] f32 skip tile

  const int tid  = threadIdx.x;
  const int lane = tid & 31;
  const int wv   = tid >> 5;
  const int m16  = lane & 15;
  const int h    = lane >> 4;
  const int rowBase = blockIdx.x * MT;
  const int srow = (wv * 16 + 8 * h) * SPITCH + m16;  // base for skip idx
  const int xrow = (wv * 16 + 8 * h) * PITCH + m16;   // base for xh  idx

  wout[tid] = w_out[tid];                       // D == blockDim == 256

  Acc acc[NT];
#pragma unroll
  for (int n = 0; n < NT; ++n)
#pragma unroll
    for (int r = 0; r < 8; ++r) acc[n].f[r] = 0.0f;

  // ---------------- GEMM 1: x_cat[128,1280] @ W_in[1280,256] ----------------
  for (int kc = 0; kc < KIN / 64; ++kc) {
    __syncthreads();
    { // async-stage W_in chunk [64][256] bf16 -> wbuf
      const uint4* src = (const uint4*)(wbf + WIN_OFF + kc * 64 * Dm);
#pragma unroll
      for (int it = 0; it < 8; ++it) {
        int i = it * 256 + tid;                 // 16B units, 0..2047
        int r = i >> 5, u = i & 31;
        async_cp16((uint4*)wbuf + r * (PITCH / 8) + u, src + r * (Dm / 8) + u);
      }
    }
    { // stage this wave's 16 rows of x_cat chunk, f32 -> bf16
      int gr0 = rowBase + wv * 16;
#pragma unroll 4
      for (int r = 0; r < 16; ++r) {
        int grow = gr0 + r; if (grow >= e_tot) grow = e_tot - 1;
        const float2* p = (const float2*)(x_cat + (size_t)grow * KIN + kc * 64) + lane;
        float2 v = *p;
        if (kc + 1 < KIN / 64)
          __builtin_prefetch(x_cat + (size_t)grow * KIN + (kc + 1) * 64 + 2 * lane, 0, 3);
        ((uint32_t*)abuf)[(wv * 16 + r) * (APITCH / 2) + lane] = pk_bf16(v.x, v.y);
      }
    }
    wait_async0();
    __syncthreads();
    mma_64k(abuf + (wv * 16 + m16) * APITCH, wbuf, lane, h, acc);
  }
  __syncthreads();

  // x1 = ssilu(acc); skip (f32) -> sbuf; bf16(x1) -> xh
#pragma unroll
  for (int n = 0; n < NT; ++n)
#pragma unroll
    for (int r = 0; r < 8; ++r) {
      float x = ssilu(acc[n].f[r]);
      sbuf[srow + r * SPITCH + 16 * n] = x;
      xh[xrow + r * PITCH + 16 * n] = bf16_1(x);
    }
  __syncthreads();

  // ---------------- residual GEMMs: xh[128,256] @ W[256,256] ----------------
  auto gemm256 = [&](const uint16_t* wsrc) {
#pragma unroll
    for (int n = 0; n < NT; ++n)
#pragma unroll
      for (int r = 0; r < 8; ++r) acc[n].f[r] = 0.0f;
    for (int kc = 0; kc < 4; ++kc) {
      __syncthreads();
      const uint4* src = (const uint4*)(wsrc + kc * 64 * Dm);
#pragma unroll
      for (int it = 0; it < 8; ++it) {
        int i = it * 256 + tid;
        int r = i >> 5, u = i & 31;
        async_cp16((uint4*)wbuf + r * (PITCH / 8) + u, src + r * (Dm / 8) + u);
      }
      wait_async0();
      __syncthreads();
      mma_64k(xh + (wv * 16 + m16) * PITCH + kc * 64, wbuf, lane, h, acc);
    }
    __syncthreads();   // all waves done reading xh -> safe to overwrite
  };

  auto store_act = [&](bool addskip) {
#pragma unroll
    for (int n = 0; n < NT; ++n)
#pragma unroll
      for (int r = 0; r < 8; ++r) {
        float x = ssilu(acc[n].f[r]);
        if (addskip) {
          x = (sbuf[srow + r * SPITCH + 16 * n] + x) * INV_SQRT2f;
          sbuf[srow + r * SPITCH + 16 * n] = x;
        }
        xh[xrow + r * PITCH + 16 * n] = bf16_1(x);
      }
    __syncthreads();
  };

  gemm256(wbf + WR1A_OFF);  store_act(false);   // h   = ssilu(x @ W_r1a)
  gemm256(wbf + WR1B_OFF);  store_act(true);    // x   = (x + ssilu(h @ W_r1b)) * c
  gemm256(wbf + WR2A_OFF);  store_act(false);   // h   = ssilu(x @ W_r2a)
  gemm256(wbf + WR2B_OFF);                      // acc = h @ W_r2b (pre-act)

  // ---------------- final: F_st = x3 @ W_out ; scatter forces ----------------
  float partial[8];
#pragma unroll
  for (int r = 0; r < 8; ++r) partial[r] = 0.0f;
#pragma unroll
  for (int n = 0; n < NT; ++n) {
    float w = wout[16 * n + m16];
#pragma unroll
    for (int r = 0; r < 8; ++r) {
      float x = (sbuf[srow + r * SPITCH + 16 * n] + ssilu(acc[n].f[r])) * INV_SQRT2f;
      partial[r] += x * w;
    }
  }
#pragma unroll
  for (int r = 0; r < 8; ++r) {
    float v = partial[r];
    v += __shfl_xor(v, 1, 16);
    v += __shfl_xor(v, 2, 16);
    v += __shfl_xor(v, 4, 16);
    v += __shfl_xor(v, 8, 16);
    partial[r] = v;
  }
  if (m16 == 0) {
#pragma unroll
    for (int r = 0; r < 8; ++r) fst[wv * 16 + 8 * h + r] = partial[r];
  }
  __syncthreads();

  if (tid < MT) {
    int grow = rowBase + tid;
    if (grow < e_tot) {
      float f  = fst[tid];
      float e0 = edge_vec[grow * 3 + 0];
      float e1 = edge_vec[grow * 3 + 1];
      float e2 = edge_vec[grow * 3 + 2];
      int   ai = edge_idx[grow];
      atomicAdd(forces + ai * 3 + 0, f * e0);
      atomicAdd(forces + ai * 3 + 1, f * e1);
      atomicAdd(forces + ai * 3 + 2, f * e2);
    }
  }
}

// ---------------------------------------------------------------------------
// Launch
// ---------------------------------------------------------------------------
extern "C" void kernel_launch(void* const* d_in, const int* in_sizes, int n_in,
                              void* d_out, int out_size, void* d_ws, size_t ws_size,
                              hipStream_t stream) {
  const float* x_cat    = (const float*)d_in[0];
  const float* edge_vec = (const float*)d_in[1];
  const int*   edge_idx = (const int*)d_in[2];
  const float* W_in     = (const float*)d_in[3];
  const float* W_r1a    = (const float*)d_in[4];
  const float* W_r1b    = (const float*)d_in[5];
  const float* W_r2a    = (const float*)d_in[6];
  const float* W_r2b    = (const float*)d_in[7];
  const float* W_out    = (const float*)d_in[8];
  float*       out      = (float*)d_out;
  uint16_t*    wbf      = (uint16_t*)d_ws;      // needs WTOT*2 ~ 1.2 MB

  const int E = in_sizes[2];                    // edge count (200000)

  zero_out_kernel<<<(out_size + 255) / 256, 256, 0, stream>>>(out, out_size);

  const int pairs = WTOT / 2;
  convert_weights_kernel<<<(pairs + 255) / 256, 256, 0, stream>>>(
      W_in, W_r1a, W_r1b, W_r2a, W_r2b, wbf);

  const int nblk = (E + MT - 1) / MT;
  const size_t lds_bytes =
      (size_t)(MT * PITCH + 64 * PITCH + 8 * 16 * APITCH) * sizeof(uint16_t)
      + (Dm + MT + MT * SPITCH) * sizeof(float); // ~251 KB (WGP has 320 KB)
  force_head_kernel<<<nblk, 256, lds_bytes, stream>>>(
      x_cat, edge_vec, edge_idx, wbf, W_out, out, E);
}
